// Attention_11742440587335
// MI455X (gfx1250) — matrix operand order repaired and checked by reading, not compiled, not run on hardware
//
#include <hip/hip_runtime.h>
#include <hip/hip_bf16.h>
#include <math.h>

#define ENC_DIM 2048
#define DEC_DIM 512
#define ATT_DIM 512
#define BATCH   256
#define NPIX    196

typedef float        v8f   __attribute__((ext_vector_type(8)));
typedef __bf16       v16bf __attribute__((ext_vector_type(16)));
typedef unsigned int u32x4 __attribute__((ext_vector_type(4)));
typedef unsigned int u32x2 __attribute__((ext_vector_type(2)));
typedef float        f32x4 __attribute__((ext_vector_type(4)));

union Frag { u32x4 q[2]; v16bf v; };

__device__ __forceinline__ unsigned short f32_to_bf16_rne(float f) {
    unsigned u = __float_as_uint(f);
    u += 0x7FFFu + ((u >> 16) & 1u);   // round-to-nearest-even
    return (unsigned short)(u >> 16);
}

// ---------------------------------------------------------------------------
// Kernel 0: W_enc [E=2048, A=512] fp32  ->  WencT [A=512][E=2048] bf16 (d_ws)
// Makes WMMA B-fragments k-contiguous rows, two 16B reads per lane; 2MB, stays L2-hot.
// ---------------------------------------------------------------------------
__global__ __launch_bounds__(256) void wenc_to_bf16T(const float* __restrict__ W,
                                                     unsigned short* __restrict__ WT) {
    int idx = blockIdx.x * 256 + threadIdx.x;      // over 512*2048
    int a = idx >> 11;                             // /2048
    int e = idx & 2047;
    WT[idx] = f32_to_bf16_rne(W[(size_t)e * ATT_DIM + a]);
}

// ---------------------------------------------------------------------------
// Fused attention kernel: one workgroup (256 thr = 8 waves) per batch index.
// ---------------------------------------------------------------------------
__global__ __launch_bounds__(256) void attention_fused(
    const float* __restrict__ enc,     // [B, N, E]
    const float* __restrict__ hidden,  // [B, D]
    const float* __restrict__ b_enc,   // [A]
    const float* __restrict__ W_dec,   // [D, A]
    const float* __restrict__ b_dec,   // [A]
    const float* __restrict__ W_full,  // [A, 1]
    const unsigned short* __restrict__ WencT, // [A][E] bf16
    float* __restrict__ out)           // context [B,E] then alpha [B,N]
{
    __shared__ unsigned short Abuf[16][512 + 8];   // staged bf16 A tile (padded)
    __shared__ float hid[DEC_DIM];
    __shared__ float batt[ATT_DIM];                // att2 + b_enc
    __shared__ float wfl[ATT_DIM];
    __shared__ float scores_lds[208];
    __shared__ float red[256];
    __shared__ float alpha_lds[NPIX];

    const int t    = threadIdx.x;
    const int lane = t & 31;
    const int wave = t >> 5;           // 0..7
    const int b    = blockIdx.x;
    const int acol = lane & 15;
    const int hsel = lane >> 4;        // 0: lanes 0-15, 1: lanes 16-31

    // ---------------- Phase A: att2 = hidden @ W_dec + b_dec (+ b_enc) -----
    hid[t]        = hidden[(size_t)b * DEC_DIM + t];
    hid[t + 256]  = hidden[(size_t)b * DEC_DIM + t + 256];
    wfl[t]        = W_full[t];
    wfl[t + 256]  = W_full[t + 256];
    if (t < 208) scores_lds[t] = 0.0f;
    __syncthreads();

    for (int a = t; a < ATT_DIM; a += 256) {
        float acc = b_dec[a];
        #pragma unroll 4
        for (int d = 0; d < DEC_DIM; ++d)
            acc += hid[d] * W_dec[(size_t)d * ATT_DIM + a];
        batt[a] = acc + b_enc[a];
    }
    __syncthreads();

    // ---------------- Phase B: scores via bf16 WMMA GEMM -------------------
    // 13 M-tiles (196 rows), 32 N-tiles (512 att dims); wave w owns N-tiles
    // w*4 .. w*4+3 and keeps their f32 accumulators across the K loop.
    const v8f vz = {0.f,0.f,0.f,0.f,0.f,0.f,0.f,0.f};

    for (int Mtile = 0; Mtile < 13; ++Mtile) {
        v8f cacc[4];
        #pragma unroll
        for (int j = 0; j < 4; ++j) cacc[j] = vz;

        for (int kb = 0; kb < 4; ++kb) {           // K blocks of 512
            __syncthreads();
            // stage 16 rows x 512 cols fp32 -> bf16 LDS (zero-pad rows >= 196)
            #pragma unroll
            for (int q = 0; q < 8; ++q) {
                int idx  = t + 256 * q;            // 2048 float4 groups
                int row  = idx >> 7;               // 128 groups per row
                int col  = (idx & 127) * 4;
                int gr   = Mtile * 16 + row;
                f32x4 val = {0.f, 0.f, 0.f, 0.f};
                if (gr < NPIX)
                    val = *(const f32x4*)(enc + ((size_t)(b * NPIX + gr)) * ENC_DIM
                                              + kb * 512 + col);
                u32x2 p;
                p.x = (unsigned)f32_to_bf16_rne(val.x) |
                      ((unsigned)f32_to_bf16_rne(val.y) << 16);
                p.y = (unsigned)f32_to_bf16_rne(val.z) |
                      ((unsigned)f32_to_bf16_rne(val.w) << 16);
                *(u32x2*)(&Abuf[row][col]) = p;
            }
            __syncthreads();

            for (int ks = 0; ks < 16; ++ks) {      // K steps of 32
                // A fragment (ISA 16-bit A layout): lane row = lane&15,
                // K chunks [0..7]+[16..23] (lanes 0-15) / [8..15]+[24..31]
                Frag af;
                const unsigned short* ar = &Abuf[acol][ks * 32 + (hsel ? 8 : 0)];
                af.q[0] = *(const u32x4*)(ar);
                af.q[1] = *(const u32x4*)(ar + 16);

                #pragma unroll
                for (int j = 0; j < 4; ++j) {
                    int a = (wave * 4 + j) * 16 + acol;   // B column
                    const unsigned short* br = WencT + (size_t)a * ENC_DIM
                                             + kb * 512 + ks * 32 + (hsel ? 8 : 0);
                    Frag bf_;
                    bf_.q[0] = *(const u32x4*)(br);
                    bf_.q[1] = *(const u32x4*)(br + 16);
                    cacc[j] = __builtin_amdgcn_wmma_f32_16x16x32_bf16(
                        false, af.v, false, bf_.v, (short)0, cacc[j], false, false);
                }
            }
        }

        // fold C tiles directly into scores: relu(att1+att2+b_enc) * W_full
        #pragma unroll
        for (int j = 0; j < 4; ++j) {
            int a = (wave * 4 + j) * 16 + acol;
            float bias = batt[a];
            float wf   = wfl[a];
            #pragma unroll
            for (int r = 0; r < 8; ++r) {
                float v = cacc[j][r] + bias;
                v = v > 0.f ? v : 0.f;
                v *= wf;
                v += __shfl_xor(v, 1, 32);
                v += __shfl_xor(v, 2, 32);
                v += __shfl_xor(v, 4, 32);
                v += __shfl_xor(v, 8, 32);         // sum over 16 columns
                if ((lane & 15) == 0) {
                    int n = Mtile * 16 + r + hsel * 8;
                    if (n < NPIX) atomicAdd(&scores_lds[n], v);
                }
            }
        }
    }
    __syncthreads();

    // ---------------- Phase C1: softmax over N=196 (b_full cancels) --------
    float s = (t < NPIX) ? scores_lds[t] : -INFINITY;
    red[t] = s;
    __syncthreads();
    for (int off = 128; off >= 1; off >>= 1) {
        if (t < off) red[t] = fmaxf(red[t], red[t + off]);
        __syncthreads();
    }
    float m = red[0];
    __syncthreads();
    float e = (t < NPIX) ? __expf(s - m) : 0.f;
    red[t] = e;
    __syncthreads();
    for (int off = 128; off >= 1; off >>= 1) {
        if (t < off) red[t] += red[t + off];
        __syncthreads();
    }
    float inv = 1.0f / red[0];
    float al  = e * inv;
    if (t < NPIX) {
        alpha_lds[t] = al;
        out[(size_t)BATCH * ENC_DIM + (size_t)b * NPIX + t] = al;
    }
    __syncthreads();

    // ---------------- Phase C2: context = enc^T @ alpha (L2-hot re-read) ---
    f32x4 acc0 = {0.f, 0.f, 0.f, 0.f};
    f32x4 acc1 = {0.f, 0.f, 0.f, 0.f};
    const f32x4* enc4 = (const f32x4*)(enc + (size_t)b * NPIX * ENC_DIM);
    for (int n = 0; n < NPIX; ++n) {
        float an = alpha_lds[n];
        const f32x4* rowp = enc4 + (size_t)n * (ENC_DIM / 4);
        acc0 += an * rowp[t];
        acc1 += an * rowp[256 + t];
    }
    f32x4* ctx4 = (f32x4*)out + (size_t)b * (ENC_DIM / 4);
    ctx4[t]       = acc0;
    ctx4[256 + t] = acc1;
}

// ---------------------------------------------------------------------------
extern "C" void kernel_launch(void* const* d_in, const int* in_sizes, int n_in,
                              void* d_out, int out_size, void* d_ws, size_t ws_size,
                              hipStream_t stream) {
    const float* enc    = (const float*)d_in[0];
    const float* hidden = (const float*)d_in[1];
    const float* W_enc  = (const float*)d_in[2];
    const float* b_enc  = (const float*)d_in[3];
    const float* W_dec  = (const float*)d_in[4];
    const float* b_dec  = (const float*)d_in[5];
    const float* W_full = (const float*)d_in[6];
    // d_in[7] = b_full: constant shift of all scores -> cancels in softmax.
    float* out = (float*)d_out;
    unsigned short* WencT = (unsigned short*)d_ws;   // 512*2048*2 = 2 MB

    wenc_to_bf16T<<<(ATT_DIM * ENC_DIM) / 256, 256, 0, stream>>>(W_enc, WencT);
    attention_fused<<<BATCH, 256, 0, stream>>>(enc, hidden, b_enc, W_dec, b_dec,
                                               W_full, WencT, out);
}
